// ALayer_v1_1_14637248545082
// MI455X (gfx1250) — matrix operand order repaired and checked by reading, not compiled, not run on hardware
//
#include <hip/hip_runtime.h>

typedef __attribute__((ext_vector_type(16))) __bf16 v16bf;
typedef __attribute__((ext_vector_type(8)))  float  v8f;

#define B_   32
#define C_   64
#define H_   128
#define W_   128
#define HW_  (H_ * W_)
#define CH1  16   // conv1 output channels

// ---------------------------------------------------------------------------
// Kernel 1: h = relu(conv3x3(x, w1))   (implicit GEMM via bf16 WMMA)
//   Block = 128 threads (4 waves, wave32). Each block handles one (b, y) row
//   half (64 pixels). Each wave computes a 16-pixel x 16-channel output tile:
//   K = 576 = 9 taps * 64 in-channels -> 18 x v_wmma_f32_16x16x32_bf16.
//   h stored bf16, layout [b][y][x][16] (M-major for the next conv).
// ---------------------------------------------------------------------------
__global__ __launch_bounds__(128)
void conv1_wmma_kernel(const float* __restrict__ x,
                       const float* __restrict__ w1,
                       __bf16* __restrict__ h)
{
    // xTile: [r(3)][xx(66)][c(64)]  (row-halo tile; c contiguous -> b128 frags)
    // wTile: [t(9)][n(16)][c(64)]   (K contiguous per (tap, outchan))
    __shared__ __bf16 xTile[3 * 66 * 64];
    __shared__ __bf16 wTile[9 * 16 * 64];

    const int tid   = threadIdx.x;
    const int xhalf = blockIdx.x & 1;
    const int y     = (blockIdx.x >> 1) & (H_ - 1);
    const int b     = blockIdx.x >> 8;          // 2 * 128 blocks per image
    const int xbase = xhalf * 64;

    // Cooperative load: input halo tile (fp32 -> bf16), coalesced along xx.
    for (int idx = tid; idx < 3 * 66 * 64; idx += 128) {
        int xx = idx % 66;
        int r  = (idx / 66) % 3;
        int c  = idx / (66 * 3);
        int yy = y + r - 1;
        int xg = xbase + xx - 1;
        float v = 0.0f;
        if (yy >= 0 && yy < H_ && xg >= 0 && xg < W_)
            v = x[(((size_t)b * C_ + c) * H_ + yy) * W_ + xg];
        xTile[(r * 66 + xx) * 64 + c] = (__bf16)v;
    }
    // Cooperative load: weights [16][64][3][3] fp32 -> bf16 [t][n][c].
    for (int idx = tid; idx < 9 * 16 * 64; idx += 128) {
        int c = idx & 63;
        int n = (idx >> 6) & 15;
        int t = idx >> 10;
        wTile[idx] = (__bf16)w1[(n * C_ + c) * 9 + t];
    }
    __syncthreads();

    const int lane = tid & 31;
    const int wv   = tid >> 5;        // wave id 0..3 -> 16-pixel M tile
    const int half = lane >> 4;       // lane group (ISA 16x16 layouts)
    const int m    = lane & 15;       // A-matrix row (pixel within tile)
    const int n    = m;               // B/D column (output channel)

    v8f acc = {};
    #pragma unroll
    for (int t = 0; t < 9; ++t) {
        const int r   = t / 3;                 // tap row
        const int dxp = t % 3;                 // tap col (+1 halo bias folded)
        const int xx  = wv * 16 + m + dxp;     // 0..65
        #pragma unroll
        for (int cb = 0; cb < C_; cb += 32) {  // K blocks of 32 channels
            v16bf a, bm;
            // A fragment (16-bit A 16x32 layout): element j -> K index
            //   k = (j%8) + (j/8)*16 + half*8
            #pragma unroll
            for (int j = 0; j < 16; ++j) {
                int k = (j & 7) + ((j >> 3) << 4) + (half << 3);
                a[j] = xTile[(r * 66 + xx) * 64 + cb + k];
            }
            // B fragment (16-bit B 32x16): element j -> K index
            //   k = (j%8) + (j/8)*16 + half*8 (same as A).
            #pragma unroll
            for (int j = 0; j < 16; ++j) {
                bm[j] = wTile[(t * 16 + n) * 64 + cb + (j & 7) + ((j >> 3) << 4) + (half << 3)];
            }
            acc = __builtin_amdgcn_wmma_f32_16x16x32_bf16(
                false, a, false, bm, (short)0, acc, false, false);
        }
    }

    // D layout: VGPR r8 holds M = r8 + 8*half, N = lane%16. ReLU + store.
    const int x0 = xbase + wv * 16;
    #pragma unroll
    for (int r8 = 0; r8 < 8; ++r8) {
        int   mm = r8 + (half << 3);
        float v  = acc[r8] > 0.0f ? acc[r8] : 0.0f;
        h[((((size_t)b * H_ + y) * W_) + (x0 + mm)) * CH1 + n] = (__bf16)v;
    }
}

// ---------------------------------------------------------------------------
// Kernel 2: A = sigmoid(conv3x3(h, w2)), 16 -> 1 channels (tiny: VALU).
// ---------------------------------------------------------------------------
__global__ __launch_bounds__(256)
void conv2_sig_kernel(const __bf16* __restrict__ h,
                      const float* __restrict__ w2,
                      float* __restrict__ A)
{
    int pix = blockIdx.x * 256 + threadIdx.x;
    if (pix >= B_ * HW_) return;
    int b   = pix >> 14;
    int rem = pix & (HW_ - 1);
    int y   = rem >> 7;
    int xc  = rem & (W_ - 1);

    float s = 0.0f;
    for (int ty = 0; ty < 3; ++ty) {
        int yy = y + ty - 1;
        if (yy < 0 || yy >= H_) continue;
        for (int tx = 0; tx < 3; ++tx) {
            int xg = xc + tx - 1;
            if (xg < 0 || xg >= W_) continue;
            const __bf16* hp = &h[(((size_t)b * H_ + yy) * W_ + xg) * CH1];
            #pragma unroll
            for (int c = 0; c < CH1; ++c)
                s += (float)hp[c] * w2[(c * 3 + ty) * 3 + tx];
        }
    }
    A[pix] = 1.0f / (1.0f + __expf(-s));
}

// ---------------------------------------------------------------------------
// Kernel 3: out = x * boxsum3x3(A). One thread per pixel, loop 64 channels.
// Streaming-bound: ~134MB x + ~134MB out; x likely resident in 192MB L2.
// ---------------------------------------------------------------------------
__global__ __launch_bounds__(256)
void box_mul_kernel(const float* __restrict__ x,
                    const float* __restrict__ A,
                    float* __restrict__ out)
{
    int pix = blockIdx.x * 256 + threadIdx.x;
    if (pix >= B_ * HW_) return;
    int b   = pix >> 14;
    int rem = pix & (HW_ - 1);
    int y   = rem >> 7;
    int xc  = rem & (W_ - 1);

    const float* Ab = A + (size_t)b * HW_;
    float box = 0.0f;
    #pragma unroll
    for (int dy = -1; dy <= 1; ++dy) {
        int yy = y + dy;
        if (yy < 0 || yy >= H_) continue;
        #pragma unroll
        for (int dx = -1; dx <= 1; ++dx) {
            int xg = xc + dx;
            if (xg < 0 || xg >= W_) continue;
            box += Ab[yy * W_ + xg];
        }
    }

    size_t base = (size_t)b * C_ * HW_ + rem;
    #pragma unroll 4
    for (int c = 0; c < C_; ++c) {
        size_t off = base + (size_t)c * HW_;
        out[off] = x[off] * box;
    }
}

// ---------------------------------------------------------------------------
extern "C" void kernel_launch(void* const* d_in, const int* in_sizes, int n_in,
                              void* d_out, int out_size, void* d_ws, size_t ws_size,
                              hipStream_t stream)
{
    const float* x  = (const float*)d_in[0];
    // d_in[1] = "weights" (dead argument in the reference)
    const float* w1 = (const float*)d_in[2];
    const float* w2 = (const float*)d_in[3];
    float* out = (float*)d_out;

    // workspace: h (bf16, 16.8 MB) then A (fp32, 2.1 MB)
    __bf16* h = (__bf16*)d_ws;
    float*  A = (float*)((char*)d_ws + (size_t)B_ * HW_ * CH1 * sizeof(__bf16));

    conv1_wmma_kernel<<<B_ * H_ * 2, 128, 0, stream>>>(x, w1, h);

    int pixBlocks = (B_ * HW_ + 255) / 256;
    conv2_sig_kernel<<<pixBlocks, 256, 0, stream>>>(h, w2, A);
    box_mul_kernel<<<pixBlocks, 256, 0, stream>>>(x, A, out);
}
